// MultiAttention_39711267619187
// MI455X (gfx1250) — matrix operand order repaired and checked by reading, not compiled, not run on hardware
//
#include <hip/hip_runtime.h>
#include <hip/hip_bf16.h>

// ---------------------------------------------------------------------------
// MI455X (gfx1250) fused transformer-block + output GEMM, bf16 WMMA.
//
// Roofline: 149 GFLOP vs 170 MB mandatory HBM traffic => compute-bound at
// bf16 WMMA rates, so: (1) all matmuls on v_wmma_f32_16x16x32_bf16,
// (2) fully fused per 16-row batch tile so the 201 MB "f" intermediate
// never leaves LDS, (3) weights pre-converted to bf16 once (L2-resident).
// ---------------------------------------------------------------------------

typedef __attribute__((ext_vector_type(16))) __bf16 v16bf;
typedef __attribute__((ext_vector_type(8)))  float  v8f;

union Frag16 { v16bf v; uint4 q[2]; };

#define WMMA_BF16(A, B, C) \
  __builtin_amdgcn_wmma_f32_16x16x32_bf16(false, (A), false, (B), (short)0, (C), false, false)

// LDS map (bytes), lifetimes overlapped:
//   m    fp32 96x256 : [0,      98304)   alive S0..S5
//   qkv  fp32 96x384 : [98304,  245760)  alive S1..S2
//   nbf  bf16 96x256 : [98304,  147456)  alive S3..S4  (reuses dead qkv)
//   nout bf16 96x256 : [147456, 196608)  alive S4..S5  (reuses dead qkv)
//   fbf  bf16 16x3072: [196608, 294912)  alive S5..S6  (reuses dead qkv tail)
#define SM_M     0
#define SM_QKV   98304
#define SM_NBF   98304
#define SM_NOUT  147456
#define SM_FBF   196608
#define SMEM_BYTES 294912

__global__ void cvt_f32_to_bf16(const float* __restrict__ s,
                                __bf16* __restrict__ d, int n) {
  int i = (blockIdx.x * blockDim.x + threadIdx.x) * 4;
  if (i + 3 < n) {
    const float4 v = *(const float4*)(s + i);
    d[i + 0] = (__bf16)v.x;
    d[i + 1] = (__bf16)v.y;
    d[i + 2] = (__bf16)v.z;
    d[i + 3] = (__bf16)v.w;
  } else {
    for (; i < n; ++i) d[i] = (__bf16)s[i];
  }
}

__global__ __launch_bounds__(256, 1) void fused_pipeline(
    const float* __restrict__ x,        // (B,6,128)
    const float* __restrict__ attn_b,   // (384,)
    const float* __restrict__ ln1_g, const float* __restrict__ ln1_b,   // (256,)
    const float* __restrict__ fb,       // (256,)
    const float* __restrict__ ln2_g, const float* __restrict__ ln2_b,   // (512,)
    const float* __restrict__ ob,       // (512,)
    const __bf16* __restrict__ awb,     // attn_w bf16 (384,128)
    const __bf16* __restrict__ fwb,     // fw bf16 (256,256)
    const __bf16* __restrict__ owb,     // ow bf16 (512,3072)
    float* __restrict__ out)            // (B,512)
{
  extern __shared__ char smem[];
  float*  sm_m    = (float*) (smem + SM_M);     // 96 x 256
  float*  sm_qkv  = (float*) (smem + SM_QKV);   // 96 x 384
  __bf16* sm_nbf  = (__bf16*)(smem + SM_NBF);   // 96 x 256
  __bf16* sm_nout = (__bf16*)(smem + SM_NOUT);  // 96 x 256
  __bf16* sm_fbf  = (__bf16*)(smem + SM_FBF);   // 16 x 3072

  const int tid  = threadIdx.x;
  const int wave = tid >> 5;
  const int lane = tid & 31;
  const int lrow = lane & 15;   // A-frag M row / B-frag N col / C-frag N col
  const int lhi  = lane >> 4;
  const int bb   = blockIdx.x * 16;   // first batch row of this tile

  // ---- S0: x tile (contiguous 96x128 fp32) -> m[:,0:128] ----------------
  {
    const float4* xg = (const float4*)(x + (size_t)bb * 768);
    float4* m4 = (float4*)sm_m;
    for (int i4 = tid; i4 < 3072; i4 += 256) {       // uniform 12 trips
      const int r = i4 >> 5, c4 = i4 & 31;
      m4[r * 64 + c4] = xg[i4];
    }
  }
  __syncthreads();

  // ---- S1: qkv(96x384) = m[:,0:128] @ awb^T + attn_b  (WMMA bf16) -------
  for (int t = wave; t < 144; t += 8) {              // 18 tiles/wave, uniform
    const int mt = t / 24, nt = t - mt * 24;
    const int ocol = nt * 16 + lrow;
    const __bf16* brow = awb + ocol * 128;
    const float*  arow = sm_m + (mt * 16 + lrow) * 256;
    v8f acc = {};
    #pragma unroll
    for (int ks = 0; ks < 4; ++ks) {
      const int ka = ks * 32 + lhi * 8;
      Frag16 af, bf;
      #pragma unroll
      for (int e = 0; e < 8; ++e) {                  // fp32 LDS -> bf16 frag
        af.v[e]     = (__bf16)arow[ka + e];
        af.v[e + 8] = (__bf16)arow[ka + 16 + e];
      }
      const int kb = ks * 32 + lhi * 8;
      bf.q[0] = *(const uint4*)(brow + kb);
      bf.q[1] = *(const uint4*)(brow + kb + 16);
      acc = WMMA_BF16(af.v, bf.v, acc);
    }
    const float bias = attn_b[ocol];
    #pragma unroll
    for (int j = 0; j < 8; ++j)
      sm_qkv[(mt * 16 + j + lhi * 8) * 384 + ocol] = acc[j] + bias;
  }
  __syncthreads();

  // ---- S2: attention core (no softmax), fp32 VALU -----------------------
  {
    const int p = tid >> 2, sub = tid & 3;           // 64 (bi,h) pairs x 4
    const int bi = p >> 2, h = p & 3;
    const float* base = sm_qkv + (bi * 6) * 384 + h * 96;
    float sc[6][6];
    #pragma unroll
    for (int qi = 0; qi < 6; ++qi) {
      const float* Q = base + qi * 384;
      #pragma unroll
      for (int ki = 0; ki < 6; ++ki) {
        const float* K = base + ki * 384 + 32;
        float d = 0.f;
        #pragma unroll
        for (int c = 0; c < 32; ++c) d += Q[c] * K[c];
        sc[qi][ki] = d;
      }
    }
    #pragma unroll
    for (int qi = 0; qi < 6; ++qi) {
      float r[8] = {0.f, 0.f, 0.f, 0.f, 0.f, 0.f, 0.f, 0.f};
      #pragma unroll
      for (int ki = 0; ki < 6; ++ki) {
        const float s = sc[qi][ki];
        const float* V = base + ki * 384 + 64 + sub * 8;
        #pragma unroll
        for (int d = 0; d < 8; ++d) r[d] += s * V[d];
      }
      float* mrow = sm_m + (bi * 6 + qi) * 256 + 128 + h * 32 + sub * 8;
      #pragma unroll
      for (int d = 0; d < 8; ++d) mrow[d] = r[d];
    }
  }
  __syncthreads();

  // ---- S3: LN1(m, 256) -> nbf (bf16) ------------------------------------
  if (tid < 96) {
    const float* row = sm_m + tid * 256;
    float mu = 0.f;
    for (int c = 0; c < 256; ++c) mu += row[c];
    mu *= (1.f / 256.f);
    float var = 0.f;
    for (int c = 0; c < 256; ++c) { const float d = row[c] - mu; var += d * d; }
    var *= (1.f / 256.f);
    const float rs = rsqrtf(var + 1e-5f);
    __bf16* o = sm_nbf + tid * 256;
    for (int c = 0; c < 256; ++c)
      o[c] = (__bf16)((row[c] - mu) * rs * ln1_g[c] + ln1_b[c]);
  }
  __syncthreads();

  // ---- S4: nout(96x256) = nbf @ fwb^T + fb  (WMMA bf16) -----------------
  for (int t = wave; t < 96; t += 8) {               // 12 tiles/wave, uniform
    const int mt = t >> 4, nt = t & 15;
    const int ocol = nt * 16 + lrow;
    const __bf16* brow = fwb + ocol * 256;
    const __bf16* arow = sm_nbf + (mt * 16 + lrow) * 256;
    v8f acc = {};
    #pragma unroll
    for (int ks = 0; ks < 8; ++ks) {
      const int ka = ks * 32 + lhi * 8;
      const int kb = ks * 32 + lhi * 8;
      Frag16 af, bf;
      af.q[0] = *(const uint4*)(arow + ka);
      af.q[1] = *(const uint4*)(arow + ka + 16);
      bf.q[0] = *(const uint4*)(brow + kb);
      bf.q[1] = *(const uint4*)(brow + kb + 16);
      acc = WMMA_BF16(af.v, bf.v, acc);
    }
    const float bias = fb[ocol];
    #pragma unroll
    for (int j = 0; j < 8; ++j)
      sm_nout[(mt * 16 + j + lhi * 8) * 256 + ocol] = (__bf16)(acc[j] + bias);
  }
  __syncthreads();

  // ---- S5: f = LN2([nout | m], 512) -> fbf (16 x 3072 bf16) -------------
  if (tid < 96) {
    const int bi = tid / 6, s = tid - bi * 6;
    const __bf16* nrow = sm_nout + tid * 256;
    const float*  mrow = sm_m + tid * 256;
    float mu = 0.f;
    for (int c = 0; c < 256; ++c) mu += (float)nrow[c];
    for (int c = 0; c < 256; ++c) mu += mrow[c];
    mu *= (1.f / 512.f);
    float var = 0.f;
    for (int c = 0; c < 256; ++c) { const float d = (float)nrow[c] - mu; var += d * d; }
    for (int c = 0; c < 256; ++c) { const float d = mrow[c] - mu;        var += d * d; }
    var *= (1.f / 512.f);
    const float rs = rsqrtf(var + 1e-5f);
    __bf16* fr = sm_fbf + bi * 3072 + s * 512;
    for (int c = 0; c < 256; ++c)
      fr[c] = (__bf16)(((float)nrow[c] - mu) * rs * ln2_g[c] + ln2_b[c]);
    for (int c = 0; c < 256; ++c)
      fr[256 + c] = (__bf16)((mrow[c] - mu) * rs * ln2_g[256 + c] + ln2_b[256 + c]);
  }
  __syncthreads();

  // ---- S6: out(16x512) = fbf(16x3072) @ owb^T + ob  (dominant GEMM) -----
  {
    v8f acc[4] = {};
    const int nt0 = wave * 4;                        // 4 N-tiles per wave
    const __bf16* arow = sm_fbf + lrow * 3072;
    const __bf16* br[4];
    #pragma unroll
    for (int i = 0; i < 4; ++i)
      br[i] = owb + (size_t)((nt0 + i) * 16 + lrow) * 3072;

    for (int ks = 0; ks < 96; ++ks) {
      const int ka = ks * 32 + lhi * 8;
      const int kb = ks * 32 + lhi * 8;
      Frag16 af;
      af.q[0] = *(const uint4*)(arow + ka);          // ds_load_b128
      af.q[1] = *(const uint4*)(arow + ka + 16);
      #pragma unroll
      for (int i = 0; i < 4; ++i) {
        Frag16 bf;
        bf.q[0] = *(const uint4*)(br[i] + kb);       // global_load_b128 (L2 hit)
        bf.q[1] = *(const uint4*)(br[i] + kb + 16);
        acc[i] = WMMA_BF16(af.v, bf.v, acc[i]);
      }
    }
    #pragma unroll
    for (int i = 0; i < 4; ++i) {
      const int o = (nt0 + i) * 16 + lrow;
      const float bias = ob[o];
      #pragma unroll
      for (int j = 0; j < 8; ++j)
        out[(size_t)(bb + j + lhi * 8) * 512 + o] = acc[i][j] + bias;
    }
  }
}

extern "C" void kernel_launch(void* const* d_in, const int* in_sizes, int n_in,
                              void* d_out, int out_size, void* d_ws, size_t ws_size,
                              hipStream_t stream) {
  (void)in_sizes; (void)n_in; (void)out_size; (void)ws_size;

  const float* x      = (const float*)d_in[0];
  const float* attn_w = (const float*)d_in[1];
  const float* attn_b = (const float*)d_in[2];
  const float* ln1_g  = (const float*)d_in[3];
  const float* ln1_b  = (const float*)d_in[4];
  const float* fw     = (const float*)d_in[5];
  const float* fb     = (const float*)d_in[6];
  const float* ln2_g  = (const float*)d_in[7];
  const float* ln2_b  = (const float*)d_in[8];
  const float* ow     = (const float*)d_in[9];
  const float* ob     = (const float*)d_in[10];
  float* out = (float*)d_out;

  // Workspace: bf16 weight copies (3.4 MB total), rebuilt every call.
  __bf16* owb = (__bf16*)d_ws;            // 512*3072
  __bf16* awb = owb + 512 * 3072;         // 384*128
  __bf16* fwb = awb + 384 * 128;          // 256*256

  cvt_f32_to_bf16<<<(512 * 3072 / 4 + 255) / 256, 256, 0, stream>>>(ow, owb, 512 * 3072);
  cvt_f32_to_bf16<<<(384 * 128  / 4 + 255) / 256, 256, 0, stream>>>(attn_w, awb, 384 * 128);
  cvt_f32_to_bf16<<<(256 * 256  / 4 + 255) / 256, 256, 0, stream>>>(fw, fwb, 256 * 256);

  (void)hipFuncSetAttribute(reinterpret_cast<const void*>(fused_pipeline),
                            hipFuncAttributeMaxDynamicSharedMemorySize, SMEM_BYTES);

  fused_pipeline<<<32768 / 16, 256, SMEM_BYTES, stream>>>(
      x, attn_b, ln1_g, ln1_b, fb, ln2_g, ln2_b, ob, awb, fwb, owb, out);
}
